// LocalAttentionCluster_74706661147120
// MI455X (gfx1250) — matrix operand order repaired and checked by reading, compiled, not run on hardware
//
#include <hip/hip_runtime.h>
#include <hip/hip_bf16.h>

typedef __bf16 bf16_t;
typedef __attribute__((ext_vector_type(16))) __bf16 v16bf;
typedef __attribute__((ext_vector_type(8)))  __bf16 v8bf;
typedef __attribute__((ext_vector_type(8)))  float  v8f;
typedef __attribute__((ext_vector_type(4))) unsigned int u32x4;
typedef __attribute__((ext_vector_type(8))) int          i32x8;
typedef __attribute__((ext_vector_type(4))) int          i32x4;

#define GB   8
#define GL   4096
#define GD   1024
#define GM   (GB * GL)     // 32768 rows
#define TILE_M 128
#define TILE_N 128
#define TILE_K 32
#define LDSS   40          // padded row stride (elements): conflict-free b128 reads

#if defined(__has_builtin)
#  if __has_builtin(__builtin_amdgcn_tensor_load_to_lds)
#    define HAVE_TDM 1
#  endif
#endif
#ifndef HAVE_TDM
#  define HAVE_TDM 0
#endif

// ---------------------------------------------------------------------------
// fp32 -> bf16 conversion (elementwise, bandwidth trivial vs GEMM traffic)
// ---------------------------------------------------------------------------
__global__ void cvt_f32_bf16(const float* __restrict__ in, bf16_t* __restrict__ out, int n) {
    int i = blockIdx.x * blockDim.x + threadIdx.x;
    int stride = gridDim.x * blockDim.x;
    for (; i < n; i += stride) out[i] = (bf16_t)in[i];
}

// Low 32 bits of a shared-aperture flat address == LDS byte offset (ISA 10.2)
__device__ __forceinline__ uint32_t lds_off(const void* p) {
    return (uint32_t)(uintptr_t)p;
}

#if HAVE_TDM
// ---------------------------------------------------------------------------
// TDM 2D tile load: global (row-major, 8-byte units) -> LDS, with per-row
// padding so LDS row stride = 64B data + 16B pad = 80B (LDSS=40 bf16).
// D# packing per CDNA5 ISA.
// ---------------------------------------------------------------------------
__device__ __forceinline__ void tdm_load_2d(uint32_t lds, uint64_t gaddr,
                                            int tensor_d0, int tensor_d1,
                                            int tile_d0, int tile_d1,
                                            int d0_stride) {
    u32x4 g0;
    g0[0] = 1u;                                       // count=1, user descriptor
    g0[1] = lds;                                      // lds_addr (bytes)
    g0[2] = (uint32_t)gaddr;                          // global_addr[31:0]
    g0[3] = (uint32_t)((gaddr >> 32) & 0x01FFFFFFu)   // global_addr[56:32]
          | 0x80000000u;                              // type=2 ("image")
    i32x8 g1;
    g1[0] = (3 << 16)    // data_size = 8 bytes
          | (1 << 20)    // pad_enable
          | (3 << 22)    // pad_interval: 16 DWORDs (=64B, one tile row)
          | (3 << 25);   // pad_amount:    4 DWORDs (=16B)  -> 80B row stride
    g1[1] = (tensor_d0 & 0xFFFF) << 16;                           // dim0[15:0]
    g1[2] = ((tensor_d0 >> 16) & 0xFFFF) | ((tensor_d1 & 0xFFFF) << 16);
    g1[3] = ((tensor_d1 >> 16) & 0xFFFF) | ((tile_d0 & 0xFFFF) << 16);
    g1[4] = tile_d1 & 0xFFFF;                                     // tile_dim2=0
    g1[5] = d0_stride;                                            // stride[31:0]
    g1[6] = 0;
    g1[7] = 0;
    i32x4 gz = {0, 0, 0, 0};
#if defined(__clang_major__) && (__clang_major__ >= 23)
    i32x8 gz8 = {0, 0, 0, 0, 0, 0, 0, 0};
    __builtin_amdgcn_tensor_load_to_lds(g0, g1, gz, gz, gz8, 0);
#else
    __builtin_amdgcn_tensor_load_to_lds(g0, g1, gz, gz, 0);
#endif
}
#endif

__device__ __forceinline__ void stage_wait() {
#if HAVE_TDM
    __builtin_amdgcn_s_wait_tensorcnt(0);    // TENSORcnt==0 already for non-issuing waves
#else
    asm volatile("s_wait_asynccnt 0" ::: "memory");
#endif
}

// ---------------------------------------------------------------------------
// C = A @ W^T  with A: MxK bf16 row-major, W: NxK bf16 row-major.
// 128x128 block tile, 8 waves (2x4), each wave 4x2 WMMA 16x16 tiles.
// Double-buffered LDS fed by the Tensor Data Mover (TENSOR_LOAD_TO_LDS,
// TENSORcnt) — fallback: GLOBAL_LOAD_ASYNC_TO_LDS_B128 (ASYNCcnt).
// v_wmma_f32_16x16x32_bf16 with f32 accumulate, OutT in {bf16, float}.
// ---------------------------------------------------------------------------
template <typename OutT>
__global__ __launch_bounds__(256) void gemm_xWT(const bf16_t* __restrict__ A,
                                                const bf16_t* __restrict__ W,
                                                OutT* __restrict__ C,
                                                int M, int N, int K) {
    __shared__ __align__(16) bf16_t As[2][TILE_M * LDSS];
    __shared__ __align__(16) bf16_t Bs[2][TILE_N * LDSS];

    const int tid   = threadIdx.x;
    const int lane  = tid & 31;           // wave32
    const int wave  = tid >> 5;           // 0..7
    const int wm    = wave >> 2;          // 0..1 : 64-row band
    const int wn    = wave & 3;           // 0..3 : 32-col band
    const int lhalf = lane >> 4;          // 0/1
    const int lmod  = lane & 15;

    const int mBase = blockIdx.y * TILE_M;
    const int nBase = blockIdx.x * TILE_N;

#if HAVE_TDM
    // One DMA op per tile, issued from wave 0 only (TDM ignores EXEC; one
    // instruction moves the whole 128x32 slab while all waves keep computing).
    auto stage = [&](int buf, int k0) {
        if (wave == 0) {
            uint64_t gA = (uint64_t)(uintptr_t)(A + (size_t)mBase * K + k0);
            uint64_t gB = (uint64_t)(uintptr_t)(W + (size_t)nBase * K + k0);
            tdm_load_2d(lds_off(&As[buf][0]), gA, K / 4, TILE_M, 8, TILE_M, K / 4);
            tdm_load_2d(lds_off(&Bs[buf][0]), gB, K / 4, TILE_N, 8, TILE_N, K / 4);
        }
    };
#else
    // Per-lane async copies: 512 16B chunks per tile, 2 per thread; SADDR form.
    auto stage = [&](int buf, int k0) {
#pragma unroll
        for (int i = 0; i < 2; ++i) {
            int c   = tid + i * 256;
            int row = c >> 2;
            int off = (c & 3) * 8;                    // bf16 elements, 16B aligned
            uint32_t ldsA = lds_off(&As[buf][row * LDSS + off]);
            uint32_t ldsB = lds_off(&Bs[buf][row * LDSS + off]);
            uint32_t gA = (uint32_t)(((unsigned)(mBase + row) * (unsigned)K + k0 + off) * 2u);
            uint32_t gB = (uint32_t)(((unsigned)(nBase + row) * (unsigned)K + k0 + off) * 2u);
            asm volatile("global_load_async_to_lds_b128 %0, %1, %2"
                         :: "v"(ldsA), "v"(gA), "s"(A) : "memory");
            asm volatile("global_load_async_to_lds_b128 %0, %1, %2"
                         :: "v"(ldsB), "v"(gB), "s"(W) : "memory");
        }
    };
#endif

    v8f acc[4][2];
#pragma unroll
    for (int mi = 0; mi < 4; ++mi)
#pragma unroll
        for (int ni = 0; ni < 2; ++ni) acc[mi][ni] = (v8f){0.f, 0.f, 0.f, 0.f, 0.f, 0.f, 0.f, 0.f};

    stage(0, 0);
    stage_wait();
    __syncthreads();

    const int nsteps = K / TILE_K;
    for (int s = 0; s < nsteps; ++s) {
        const int cur = s & 1;
        if (s + 1 < nsteps) stage(cur ^ 1, (s + 1) * TILE_K);   // prefetch next slab

        // A fragments (16x32): VGPR0-3 <- K in [8*half, +8), VGPR4-7 <- K in [16+8*half, +8)
        v16bf afrag[4];
#pragma unroll
        for (int mi = 0; mi < 4; ++mi) {
            const bf16_t* base = &As[cur][(wm * 64 + mi * 16 + lmod) * LDSS];
            v8bf lo = *reinterpret_cast<const v8bf*>(base + lhalf * 8);
            v8bf hi = *reinterpret_cast<const v8bf*>(base + 16 + lhalf * 8);
            afrag[mi] = __builtin_shufflevector(lo, hi, 0, 1, 2, 3, 4, 5, 6, 7,
                                                        8, 9, 10, 11, 12, 13, 14, 15);
        }
        // B fragments (32x16): lane holds column n = lmod, K in [8*half, +8) and [16+8*half, +8)
        v16bf bfrag[2];
#pragma unroll
        for (int ni = 0; ni < 2; ++ni) {
            const bf16_t* base = &Bs[cur][(wn * 32 + ni * 16 + lmod) * LDSS];
            v8bf lo = *reinterpret_cast<const v8bf*>(base + lhalf * 8);
            v8bf hi = *reinterpret_cast<const v8bf*>(base + 16 + lhalf * 8);
            bfrag[ni] = __builtin_shufflevector(lo, hi, 0, 1, 2, 3, 4, 5, 6, 7,
                                                        8, 9, 10, 11, 12, 13, 14, 15);
        }

#pragma unroll
        for (int mi = 0; mi < 4; ++mi)
#pragma unroll
            for (int ni = 0; ni < 2; ++ni)
                acc[mi][ni] = __builtin_amdgcn_wmma_f32_16x16x32_bf16(
                    false, afrag[mi], false, bfrag[ni], (short)0, acc[mi][ni], false, false);

        // staging into buf^1 complete; barrier publishes it and guarantees all
        // waves finished reading buf before it gets re-staged next iteration
        stage_wait();
        __syncthreads();
    }

    // C layout: VGPR i -> M = base + 8*(lane/16) + i, N = base + lane%16
#pragma unroll
    for (int mi = 0; mi < 4; ++mi) {
#pragma unroll
        for (int ni = 0; ni < 2; ++ni) {
            int col     = nBase + wn * 32 + ni * 16 + lmod;
            int rowBase = mBase + wm * 64 + mi * 16 + lhalf * 8;
#pragma unroll
            for (int i = 0; i < 8; ++i)
                C[(size_t)(rowBase + i) * N + col] = (OutT)acc[mi][ni][i];
        }
    }
}

// ---------------------------------------------------------------------------
// Cluster-local gated attention (no softmax in reference!):
//   s = (q·k) / 8 ;  w = s * sigmoid(2s) ;  masked (upper-tri) -> -10000
//   attended = w @ v   (8x8 @ 8x64 per (b, cluster, head))
// One 64-thread block per (b, n, h). ~1 GFLOP total -> VALU is fine.
// ---------------------------------------------------------------------------
__global__ __launch_bounds__(64) void attn_cluster(const bf16_t* __restrict__ Q,
                                                   const bf16_t* __restrict__ Kp,
                                                   const bf16_t* __restrict__ V,
                                                   bf16_t* __restrict__ Out) {
    __shared__ float qs[8][64], ks[8][64], vs[8][64], ws[8][8];
    const int t   = threadIdx.x;            // 0..63 : one head column per thread
    const int gid = blockIdx.x;
    const int h   = gid & 15;
    const int n   = (gid >> 4) & 511;
    const int b   = gid >> 13;
    const size_t row0 = (size_t)b * GL + (size_t)n * 8;
    const int    col0 = h * 64;

#pragma unroll
    for (int r = 0; r < 8; ++r) {
        size_t idx = (row0 + r) * GD + col0 + t;
        qs[r][t] = (float)Q[idx];
        ks[r][t] = (float)Kp[idx];
        vs[r][t] = (float)V[idx];
    }
    __syncthreads();

    {   // 64 threads == 64 (q,k) score entries
        const int qi = t >> 3, ki = t & 7;
        float s = 0.f;
#pragma unroll
        for (int d = 0; d < 64; ++d) s += qs[qi][d] * ks[ki][d];
        s *= 0.125f;                                  // 1/sqrt(64)
        float gate = 1.0f / (1.0f + __expf(-2.0f * s));
        ws[qi][ki] = (ki <= qi) ? s * gate : -10000.0f;
    }
    __syncthreads();

#pragma unroll
    for (int qi = 0; qi < 8; ++qi) {
        float acc = 0.f;
#pragma unroll
        for (int ki = 0; ki < 8; ++ki) acc += ws[qi][ki] * vs[ki][t];
        Out[(row0 + qi) * GD + col0 + t] = (bf16_t)acc;
    }
}

// ---------------------------------------------------------------------------
extern "C" void kernel_launch(void* const* d_in, const int* in_sizes, int n_in,
                              void* d_out, int out_size, void* d_ws, size_t ws_size,
                              hipStream_t stream) {
    (void)in_sizes; (void)n_in; (void)out_size; (void)ws_size;

    const float* x  = (const float*)d_in[0];
    // d_in[1] = causal_mask (unused: mask is applied analytically per cluster)
    const float* Wq = (const float*)d_in[2];
    const float* Wk = (const float*)d_in[3];
    const float* Wv = (const float*)d_in[4];
    const float* Wo = (const float*)d_in[5];
    float* out = (float*)d_out;

    char* ws = (char*)d_ws;
    const size_t XB = (size_t)GM * GD * sizeof(bf16_t);   // 64 MB
    const size_t WB = (size_t)GD * GD * sizeof(bf16_t);   //  2 MB
    bf16_t* xb  = (bf16_t*)(ws);             // x (bf16); later reused for attended
    bf16_t* qb  = (bf16_t*)(ws + XB);
    bf16_t* kb  = (bf16_t*)(ws + 2 * XB);
    bf16_t* vb  = (bf16_t*)(ws + 3 * XB);
    bf16_t* wqb = (bf16_t*)(ws + 4 * XB);
    bf16_t* wkb = (bf16_t*)(ws + 4 * XB + WB);
    bf16_t* wvb = (bf16_t*)(ws + 4 * XB + 2 * WB);
    bf16_t* wob = (bf16_t*)(ws + 4 * XB + 3 * WB);

    const int NX = GM * GD;      // 33.5M
    const int NW = GD * GD;      // 1M
    cvt_f32_bf16<<<4096, 256, 0, stream>>>(x, xb, NX);
    cvt_f32_bf16<<<1024, 256, 0, stream>>>(Wq, wqb, NW);
    cvt_f32_bf16<<<1024, 256, 0, stream>>>(Wk, wkb, NW);
    cvt_f32_bf16<<<1024, 256, 0, stream>>>(Wv, wvb, NW);
    cvt_f32_bf16<<<1024, 256, 0, stream>>>(Wo, wob, NW);

    dim3 grid(GD / TILE_N, GM / TILE_M);   // (8, 256)
    gemm_xWT<bf16_t><<<grid, 256, 0, stream>>>(xb, wqb, qb, GM, GD, GD);
    gemm_xWT<bf16_t><<<grid, 256, 0, stream>>>(xb, wkb, kb, GM, GD, GD);
    gemm_xWT<bf16_t><<<grid, 256, 0, stream>>>(xb, wvb, vb, GM, GD, GD);

    attn_cluster<<<GB * (GL / 8) * 16, 64, 0, stream>>>(qb, kb, vb, xb);

    gemm_xWT<float><<<grid, 256, 0, stream>>>(xb, wob, out, GM, GD, GD);
}
